// GNN_Model_48773648613980
// MI455X (gfx1250) — matrix operand order repaired and checked by reading, not compiled, not run on hardware
//
#include <hip/hip_runtime.h>
#include <hip/hip_bf16.h>

// ---------------------------------------------------------------------------
// RouteNet-style GNN for MI455X (gfx1250, wave32, WMMA f32_16x16x32_f16)
// ---------------------------------------------------------------------------
#define NLINKS 10000
#define NPATHS 50000
#define TT     8
#define EE     (NPATHS * TT)

typedef _Float16 h16;
typedef __attribute__((ext_vector_type(8)))  _Float16 v8h;
typedef __attribute__((ext_vector_type(16))) _Float16 v16h;
typedef __attribute__((ext_vector_type(8)))  float    v8f;
typedef __attribute__((ext_vector_type(4)))  unsigned v4u;
typedef __attribute__((ext_vector_type(4)))  int      v4i;

#if defined(__has_builtin)
#if __has_builtin(__builtin_amdgcn_global_load_async_to_lds_b128)
#define HAVE_ASYNC_LDS 1
typedef __attribute__((address_space(1))) v4i v4i_g;  // global
typedef __attribute__((address_space(3))) v4i v4i_s;  // LDS
#endif
#endif

__device__ __forceinline__ v8f splat8(float x) {
  v8f v = {x, x, x, x, x, x, x, x};
  return v;
}

// D = A(16x32 f16) * B(32x16 f16) + C(16x16 f32)
__device__ __forceinline__ v8f wmma32(v16h a, v16h b, v8f c) {
  return __builtin_amdgcn_wmma_f32_16x16x32_f16(
      /*neg_a=*/false, a, /*neg_b=*/false, b,
      /*c_mod=*/(short)0, c, /*reuse_a=*/false, /*reuse_b=*/false);
}

// A-fragment: 16x32 f16 tile, row-major with leading dim ld.
// Lane m/hs=0 holds K={0..7,16..23}; lanes 16..31 hold K={8..15,24..31}.
__device__ __forceinline__ v16h load_fragA(const h16* base, int ld) {
  const int lane = threadIdx.x & 31;
  const int m = lane & 15, hs = lane >> 4;
  const h16* p = base + m * ld + hs * 8;
  v8h lo = *(const v8h*)(p);
  v8h hi = *(const v8h*)(p + 16);
  v16h r;
#pragma unroll
  for (int i = 0; i < 8; ++i) { r[i] = lo[i]; r[i + 8] = hi[i]; }
  return r;
}

// B-fragment: B is 32x16 (KxN); W^T stored [N][K] row-major.
// Lanes 0-15 hold K={0..7,16..23} of column n, lanes 16-31 hold K={8..15,24..31}.
__device__ __forceinline__ v16h load_fragB(const h16* baseT, int ld) {
  const int lane = threadIdx.x & 31;
  const int n = lane & 15, hs = lane >> 4;
  const h16* p = baseT + n * ld + hs * 8;
  v8h lo = *(const v8h*)(p);
  v8h hi = *(const v8h*)(p + 16);
  v16h r;
#pragma unroll
  for (int i = 0; i < 8; ++i) { r[i] = lo[i]; r[i + 8] = hi[i]; }
  return r;
}

__device__ __forceinline__ float sigmoidf_(float x) {
  return __fdividef(1.0f, 1.0f + __expf(-x));
}
__device__ __forceinline__ float tanhf_(float x) {
  float e = __expf(2.0f * x);
  return 1.0f - __fdividef(2.0f, e + 1.0f);
}

// Monotonic float <-> uint key (for hardware integer atomic max/min).
__device__ __forceinline__ unsigned fkey(float f) {
  unsigned u = __float_as_uint(f);
  return (u & 0x80000000u) ? ~u : (u | 0x80000000u);
}
__device__ __forceinline__ float funkey(unsigned u) {
  unsigned b = (u & 0x80000000u) ? (u ^ 0x80000000u) : ~u;
  return __uint_as_float(b);
}
#define MAXKEY_INIT 0x007FFFFFu  /* fkey(-inf) */
#define MINKEY_INIT 0xFF800000u  /* fkey(+inf) */

// ---------------------------------------------------------------------------
// One-time prep: convert weights to f16, transposed [N][K]
// ---------------------------------------------------------------------------
__global__ void prep_weights(
    const float* pwx, const float* pwh, const float* lwx, const float* lwh,
    const float* aw1, const float* aw2, const float* aw3,
    const float* rw1, const float* rw2,
    h16* pwxT, h16* pwhT, h16* lwxT, h16* lwhT,
    h16* aw1T, h16* aw2T, h16* aw3T, h16* rw1T, h16* rw2T) {
  const int tid = blockIdx.x * blockDim.x + threadIdx.x;
  const int stride = gridDim.x * blockDim.x;
  auto tr = [&](const float* src, h16* dst, int K, int N) {
    for (int i = tid; i < K * N; i += stride) {
      int n = i / K, k = i % K;
      dst[n * K + k] = (h16)src[k * N + n];
    }
  };
  tr(pwx, pwxT, 32, 96);
  tr(pwh, pwhT, 32, 96);
  tr(lwx, lwxT, 32, 96);
  tr(lwh, lwhT, 32, 96);
  tr(aw1, aw1T, 128, 256);
  tr(aw2, aw2T, 256, 256);
  tr(aw3, aw3T, 256, 32);
  tr(rw1, rw1T, 32, 256);
  tr(rw2, rw2T, 256, 256);
}

// ---------------------------------------------------------------------------
// Embeddings (tiny 2->16->32 MLPs; scalar VALU is fine here)
// ---------------------------------------------------------------------------
__global__ void embed_kernel(const float* __restrict__ x0v,
                             const float* __restrict__ x1v,
                             const float* __restrict__ w1, const float* __restrict__ b1,
                             const float* __restrict__ w2, const float* __restrict__ b2,
                             float* __restrict__ out32, h16* __restrict__ out16,
                             int n) {
  int i = blockIdx.x * blockDim.x + threadIdx.x;
  if (i >= n) return;
  float x0 = x0v[i], x1 = x1v[i];
  float h1[16];
#pragma unroll
  for (int k = 0; k < 16; ++k)
    h1[k] = fmaxf(x0 * w1[k] + x1 * w1[16 + k] + b1[k], 0.f);
#pragma unroll 4
  for (int o = 0; o < 32; ++o) {
    float s = b2[o];
#pragma unroll
    for (int k = 0; k < 16; ++k) s += h1[k] * w2[k * 32 + o];
    s = fmaxf(s, 0.f);
    out32[i * 32 + o] = s;
    if (out16) out16[i * 32 + o] = (h16)s;
  }
}

__global__ void init_seg(float* segsum, unsigned* segmax, unsigned* segmin,
                         float* cnt) {
  int i = blockIdx.x * blockDim.x + threadIdx.x;
  if (i < NLINKS * 32) {
    segsum[i] = 0.f;
    segmax[i] = MAXKEY_INIT;
    segmin[i] = MINKEY_INIT;
  }
  if (i < NLINKS) cnt[i] = 0.f;
}

__global__ void count_links(const int* __restrict__ seq_links, float* cnt) {
  int e = blockIdx.x * blockDim.x + threadIdx.x;
  if (e < EE) atomicAdd(&cnt[seq_links[e]], 1.f);
}

// ---------------------------------------------------------------------------
// Fused path kernel: gather link states directly into A-fragments ->
// 8-step GRU scan (WMMA) -> segment sum/max/min atomics.
// 16 paths per wave, 8 waves per block.
// ---------------------------------------------------------------------------
__global__ void __launch_bounds__(256) path_gru_kernel(
    const h16* __restrict__ lsf16,   // [NLINKS][32]
    const float* __restrict__ ips,   // [NPATHS][32]
    const int* __restrict__ l2p,     // [EE]
    const h16* __restrict__ wxT, const h16* __restrict__ whT,  // [96][32] f16
    const float* __restrict__ bi, const float* __restrict__ br,
    float* __restrict__ segsum, unsigned* __restrict__ segmax,
    unsigned* __restrict__ segmin) {
  __shared__ alignas(16) h16 s_wxT[96 * 32];
  __shared__ alignas(16) h16 s_whT[96 * 32];
  __shared__ float s_bi[96], s_br[96];
  __shared__ alignas(16) h16 s_h[8][16 * 32];

#if HAVE_ASYNC_LDS
  // Async global->LDS copy of the GRU weights (tracked by ASYNCcnt).
  for (int i = threadIdx.x * 8; i < 96 * 32; i += blockDim.x * 8) {
    __builtin_amdgcn_global_load_async_to_lds_b128(
        (v4i_g*)(wxT + i), (v4i_s*)(s_wxT + i), 0, 0);
    __builtin_amdgcn_global_load_async_to_lds_b128(
        (v4i_g*)(whT + i), (v4i_s*)(s_whT + i), 0, 0);
  }
#if __has_builtin(__builtin_amdgcn_s_wait_asynccnt)
  __builtin_amdgcn_s_wait_asynccnt(0);
#else
  asm volatile("s_wait_asynccnt 0x0" ::: "memory");
#endif
#else
  for (int i = threadIdx.x; i < 96 * 32; i += blockDim.x) {
    s_wxT[i] = wxT[i];
    s_whT[i] = whT[i];
  }
#endif
  for (int i = threadIdx.x; i < 96; i += blockDim.x) {
    s_bi[i] = bi[i];
    s_br[i] = br[i];
  }
  __syncthreads();

  const int wave = threadIdx.x >> 5;
  const int lane = threadIdx.x & 31;
  const int m16 = lane & 15, hs = lane >> 4;
  const int pbase = (blockIdx.x * 8 + wave) * 16;
  if (pbase >= NPATHS) return;  // NPATHS % 16 == 0 -> always full tiles

  h16* hb = &s_h[wave][0];

  // h0 in C-layout registers + f16 copy in LDS for the A-fragment.
  float hreg[2][8];
#pragma unroll
  for (int k = 0; k < 2; ++k)
#pragma unroll
    for (int j = 0; j < 8; ++j) {
      int m = j + 8 * hs;
      float v = ips[(pbase + m) * 32 + k * 16 + m16];
      hreg[k][j] = v;
      hb[m * 32 + k * 16 + m16] = (h16)v;
    }

  for (int t = 0; t < TT; ++t) {
    // Gather the X A-fragment straight from global memory: this lane's
    // fragment is two contiguous 16B chunks of row m16's link state.
    int lnkA = l2p[(pbase + m16) * TT + t];
    const h16* rp = lsf16 + lnkA * 32 + hs * 8;
    v8h lo = *(const v8h*)(rp);
    v8h hi = *(const v8h*)(rp + 16);
    v16h ax;
#pragma unroll
    for (int i = 0; i < 8; ++i) { ax[i] = lo[i]; ax[i + 8] = hi[i]; }

    // Row-nonzero mask via bitwise OR (states are post-ReLU, no -0).
    v4u ou = __builtin_bit_cast(v4u, lo) | __builtin_bit_cast(v4u, hi);
    unsigned orv = ou[0] | ou[1] | ou[2] | ou[3];
    unsigned bal = (unsigned)__ballot(orv != 0u);   // wave32: low 32 bits
    unsigned rowmask = (bal | (bal >> 16)) & 0xFFFFu;

    v16h ah = load_fragA(hb, 32);
    v8f accx[6], acch[6];
#pragma unroll
    for (int nt = 0; nt < 6; ++nt) {
      accx[nt] = wmma32(ax, load_fragB(s_wxT + nt * 16 * 32, 32),
                        splat8(s_bi[nt * 16 + m16]));
      acch[nt] = wmma32(ah, load_fragB(s_whT + nt * 16 * 32, 32),
                        splat8(s_br[nt * 16 + m16]));
    }

#pragma unroll
    for (int k = 0; k < 2; ++k) {
#pragma unroll
      for (int j = 0; j < 8; ++j) {
        int c = k * 16 + m16;
        int m = j + 8 * hs;
        float z = sigmoidf_(accx[k][j] + acch[k][j]);
        float r = sigmoidf_(accx[2 + k][j] + acch[2 + k][j]);
        float hc = tanhf_(accx[4 + k][j] + r * acch[4 + k][j]);
        float hn = z * hreg[k][j] + (1.f - z) * hc;
        if ((rowmask >> m) & 1u) hreg[k][j] = hn;
        hb[m * 32 + c] = (h16)hreg[k][j];

        // Fused segment reductions (path_gather -> per-link sum/max/min).
        // Row m's link index lives in lane m (it gathered that row).
        int lnk = __shfl(lnkA, m);
        float val = hreg[k][j];
        atomicAdd(&segsum[lnk * 32 + c], val);
        unsigned kk = fkey(val);
        atomicMax(&segmax[lnk * 32 + c], kk);
        atomicMin(&segmin[lnk * 32 + c], kk);
      }
    }
  }
}

// ---------------------------------------------------------------------------
// Link update: mlp_in(128) -> 256 -> 256 -> 32 (WMMA) -> link GRU (WMMA).
// 16 links per wave, 4 waves per block.  Resets segment buffers for the
// next iteration after consuming them (each link owned by exactly one wave).
// ---------------------------------------------------------------------------
__global__ void __launch_bounds__(128) link_update_kernel(
    float* __restrict__ lstate, h16* __restrict__ lsf16,
    float* __restrict__ segsum, unsigned* __restrict__ segmax,
    unsigned* __restrict__ segmin, const float* __restrict__ cnt,
    const h16* __restrict__ w1T, const float* __restrict__ b1,
    const h16* __restrict__ w2T, const float* __restrict__ b2,
    const h16* __restrict__ w3T, const float* __restrict__ b3,
    const h16* __restrict__ gwxT, const h16* __restrict__ gwhT,
    const float* __restrict__ gbi, const float* __restrict__ gbr) {
  __shared__ alignas(16) h16 s_ain[4][16 * 128];
  __shared__ alignas(16) h16 s_a1[4][16 * 256];
  __shared__ alignas(16) h16 s_a2[4][16 * 256];
  __shared__ alignas(16) h16 s_pa[4][16 * 32];

  const int wave = threadIdx.x >> 5;
  const int lane = threadIdx.x & 31;
  const int m16 = lane & 15, hs = lane >> 4;
  const int lbase = (blockIdx.x * 4 + wave) * 16;
  if (lbase >= NLINKS) return;  // NLINKS % 16 == 0 -> full tiles

  h16* ain = s_ain[wave];
  h16* a1 = s_a1[wave];
  h16* a2 = s_a2[wave];
  h16* pa = s_pa[wave];

  // Build mlp_in = [sum | mean | max | min], and reset segment buffers.
  for (int idx = lane; idx < 16 * 32; idx += 32) {
    int r = idx >> 5, c = idx & 31;
    int l = lbase + r;
    float s = segsum[l * 32 + c];
    float cn = cnt[l];
    float mean = __fdividef(s, fmaxf(cn, 1.f));
    float mx = (cn > 0.f) ? funkey(segmax[l * 32 + c]) : 0.f;
    float mn = (cn > 0.f) ? funkey(segmin[l * 32 + c]) : 0.f;
    ain[r * 128 + c] = (h16)s;
    ain[r * 128 + 32 + c] = (h16)mean;
    ain[r * 128 + 64 + c] = (h16)mx;
    ain[r * 128 + 96 + c] = (h16)mn;
    segsum[l * 32 + c] = 0.f;
    segmax[l * 32 + c] = MAXKEY_INIT;
    segmin[l * 32 + c] = MINKEY_INIT;
  }

  // GEMM1: [16x128] @ [128x256] + b1, relu  (A-fragments hoisted)
  {
    v16h aF[4];
#pragma unroll
    for (int kc = 0; kc < 4; ++kc) aF[kc] = load_fragA(ain + kc * 32, 128);
    for (int nt = 0; nt < 16; ++nt) {
      v8f acc = splat8(b1[nt * 16 + m16]);
#pragma unroll
      for (int kc = 0; kc < 4; ++kc)
        acc = wmma32(aF[kc], load_fragB(w1T + (nt * 16) * 128 + kc * 32, 128),
                     acc);
#pragma unroll
      for (int j = 0; j < 8; ++j)
        a1[(j + 8 * hs) * 256 + nt * 16 + m16] = (h16)fmaxf(acc[j], 0.f);
    }
  }

  // GEMM2: [16x256] @ [256x256] + b2, relu  (w2T stays L2-resident)
  {
    v16h aF[8];
#pragma unroll
    for (int kc = 0; kc < 8; ++kc) aF[kc] = load_fragA(a1 + kc * 32, 256);
    for (int nt = 0; nt < 16; ++nt) {
      if (nt + 1 < 16) __builtin_prefetch(w2T + (nt + 1) * 16 * 256, 0, 3);
      v8f acc = splat8(b2[nt * 16 + m16]);
#pragma unroll
      for (int kc = 0; kc < 8; ++kc)
        acc = wmma32(aF[kc], load_fragB(w2T + (nt * 16) * 256 + kc * 32, 256),
                     acc);
#pragma unroll
      for (int j = 0; j < 8; ++j)
        a2[(j + 8 * hs) * 256 + nt * 16 + m16] = (h16)fmaxf(acc[j], 0.f);
    }
  }

  // GEMM3: [16x256] @ [256x32] + b3, relu -> pa
  {
    v16h aF[8];
#pragma unroll
    for (int kc = 0; kc < 8; ++kc) aF[kc] = load_fragA(a2 + kc * 32, 256);
#pragma unroll
    for (int nt = 0; nt < 2; ++nt) {
      v8f acc = splat8(b3[nt * 16 + m16]);
#pragma unroll
      for (int kc = 0; kc < 8; ++kc)
        acc = wmma32(aF[kc], load_fragB(w3T + (nt * 16) * 256 + kc * 32, 256),
                     acc);
#pragma unroll
      for (int j = 0; j < 8; ++j)
        pa[(j + 8 * hs) * 32 + nt * 16 + m16] = (h16)fmaxf(acc[j], 0.f);
    }
  }

  // Link GRU: x = pa, h = link_state
  v16h apa = load_fragA(pa, 32);
  v16h ah = load_fragA(lsf16 + lbase * 32, 32);
  v8f ax6[6], ah6[6];
#pragma unroll
  for (int nt = 0; nt < 6; ++nt) {
    ax6[nt] = wmma32(apa, load_fragB(gwxT + nt * 16 * 32, 32),
                     splat8(gbi[nt * 16 + m16]));
    ah6[nt] = wmma32(ah, load_fragB(gwhT + nt * 16 * 32, 32),
                     splat8(gbr[nt * 16 + m16]));
  }
#pragma unroll
  for (int k = 0; k < 2; ++k) {
#pragma unroll
    for (int j = 0; j < 8; ++j) {
      int c = k * 16 + m16;
      int m = j + 8 * hs;
      float hold = lstate[(lbase + m) * 32 + c];
      float z = sigmoidf_(ax6[k][j] + ah6[k][j]);
      float r = sigmoidf_(ax6[2 + k][j] + ah6[2 + k][j]);
      float hc = tanhf_(ax6[4 + k][j] + r * ah6[4 + k][j]);
      float hn = z * hold + (1.f - z) * hc;
      lstate[(lbase + m) * 32 + c] = hn;
      lsf16[(lbase + m) * 32 + c] = (h16)hn;
    }
  }
}

// ---------------------------------------------------------------------------
// Readout: 32 -> 256 -> 256 -> 1 per link (WMMA + scalar dot + sigmoid)
// ---------------------------------------------------------------------------
__global__ void __launch_bounds__(128) readout_kernel(
    const h16* __restrict__ lsf16,
    const h16* __restrict__ rw1T, const float* __restrict__ rb1,
    const h16* __restrict__ rw2T, const float* __restrict__ rb2,
    const float* __restrict__ rw3, const float* __restrict__ rb3,
    float* __restrict__ occ) {
  __shared__ alignas(16) h16 s_o1[4][16 * 256];
  __shared__ alignas(16) h16 s_o2[4][16 * 256];

  const int wave = threadIdx.x >> 5;
  const int lane = threadIdx.x & 31;
  const int m16 = lane & 15, hs = lane >> 4;
  const int lbase = (blockIdx.x * 4 + wave) * 16;
  if (lbase >= NLINKS) return;

  h16* o1 = s_o1[wave];
  h16* o2 = s_o2[wave];

  v16h a = load_fragA(lsf16 + lbase * 32, 32);
  for (int nt = 0; nt < 16; ++nt) {
    v8f acc = wmma32(a, load_fragB(rw1T + nt * 16 * 32, 32),
                     splat8(rb1[nt * 16 + m16]));
#pragma unroll
    for (int j = 0; j < 8; ++j)
      o1[(j + 8 * hs) * 256 + nt * 16 + m16] = (h16)fmaxf(acc[j], 0.f);
  }
  {
    v16h aF[8];
#pragma unroll
    for (int kc = 0; kc < 8; ++kc) aF[kc] = load_fragA(o1 + kc * 32, 256);
    for (int nt = 0; nt < 16; ++nt) {
      v8f acc = splat8(rb2[nt * 16 + m16]);
#pragma unroll
      for (int kc = 0; kc < 8; ++kc)
        acc = wmma32(aF[kc], load_fragB(rw2T + (nt * 16) * 256 + kc * 32, 256),
                     acc);
#pragma unroll
      for (int j = 0; j < 8; ++j)
        o2[(j + 8 * hs) * 256 + nt * 16 + m16] = (h16)fmaxf(acc[j], 0.f);
    }
  }
  // Final 256 -> 1 dot: half-lane pairs split the reduction.
  float s = 0.f;
  for (int i = hs * 128; i < hs * 128 + 128; ++i)
    s += (float)o2[m16 * 256 + i] * rw3[i];
  s += __shfl_xor(s, 16);
  if (hs == 0) occ[lbase + m16] = sigmoidf_(s + rb3[0]);
}

// ---------------------------------------------------------------------------
// Final delay: per-path sum over the T links
// ---------------------------------------------------------------------------
__global__ void final_delay_kernel(const float* __restrict__ occ,
                                   const float* __restrict__ capacity,
                                   const float* __restrict__ scale,
                                   const int* __restrict__ l2p,
                                   float* __restrict__ out) {
  int p = blockIdx.x * blockDim.x + threadIdx.x;
  if (p >= NPATHS) return;
  float qd = 0.f, td = 0.f;
#pragma unroll
  for (int t = 0; t < TT; ++t) {
    int l = l2p[p * TT + t];
    float rs = scale[l] * 5.77f + 10.5f;
    float rc = (capacity[l] * 24631.01f + 21166.35f) * rs;
    qd += occ[l] * 32.0f * 1000.0f / rc;
    td += 1000.0f / rc;
  }
  out[p] = qd + td;
}

// ---------------------------------------------------------------------------
// Host launcher
// ---------------------------------------------------------------------------
extern "C" void kernel_launch(void* const* d_in, const int* in_sizes, int n_in,
                              void* d_out, int out_size, void* d_ws,
                              size_t ws_size, hipStream_t stream) {
  // setup_inputs() dict order, params flattened in insertion order:
  const float* traffic = (const float*)d_in[0];
  const float* packets = (const float*)d_in[1];
  const float* capacity = (const float*)d_in[2];
  const float* scalev = (const float*)d_in[3];
  const int* l2p = (const int*)d_in[4];
  // d_in[5] path_ids, d_in[6] sequence_path: implicit (e = p*T + t)
  const int* seq_links = (const int*)d_in[7];
  // d_in[8] n_links, d_in[9] n_paths: compile-time constants
  const float* le_w1 = (const float*)d_in[10];
  const float* le_b1 = (const float*)d_in[11];
  const float* le_w2 = (const float*)d_in[12];
  const float* le_b2 = (const float*)d_in[13];
  const float* pe_w1 = (const float*)d_in[14];
  const float* pe_b1 = (const float*)d_in[15];
  const float* pe_w2 = (const float*)d_in[16];
  const float* pe_b2 = (const float*)d_in[17];
  const float* am_w1 = (const float*)d_in[18];
  const float* am_b1 = (const float*)d_in[19];
  const float* am_w2 = (const float*)d_in[20];
  const float* am_b2 = (const float*)d_in[21];
  const float* am_w3 = (const float*)d_in[22];
  const float* am_b3 = (const float*)d_in[23];
  const float* ro_w1 = (const float*)d_in[24];
  const float* ro_b1 = (const float*)d_in[25];
  const float* ro_w2 = (const float*)d_in[26];
  const float* ro_b2 = (const float*)d_in[27];
  const float* ro_w3 = (const float*)d_in[28];
  const float* ro_b3 = (const float*)d_in[29];
  const float* pgru_wx = (const float*)d_in[30];
  const float* pgru_wh = (const float*)d_in[31];
  const float* pgru_bi = (const float*)d_in[32];
  const float* pgru_br = (const float*)d_in[33];
  const float* lgru_wx = (const float*)d_in[34];
  const float* lgru_wh = (const float*)d_in[35];
  const float* lgru_bi = (const float*)d_in[36];
  const float* lgru_br = (const float*)d_in[37];

  // Workspace carve
  char* w = (char*)d_ws;
  auto carve = [&](size_t bytes) -> void* {
    void* p = (void*)w;
    w += (bytes + 255) & ~(size_t)255;
    return p;
  };
  float* lstate = (float*)carve((size_t)NLINKS * 32 * 4);
  h16* lsf16 = (h16*)carve((size_t)NLINKS * 32 * 2);
  float* ips = (float*)carve((size_t)NPATHS * 32 * 4);
  float* segsum = (float*)carve((size_t)NLINKS * 32 * 4);
  unsigned* segmax = (unsigned*)carve((size_t)NLINKS * 32 * 4);
  unsigned* segmin = (unsigned*)carve((size_t)NLINKS * 32 * 4);
  float* cnt = (float*)carve((size_t)NLINKS * 4);
  float* occ = (float*)carve((size_t)NLINKS * 4);
  h16* pwxT = (h16*)carve(96 * 32 * 2);
  h16* pwhT = (h16*)carve(96 * 32 * 2);
  h16* lwxT = (h16*)carve(96 * 32 * 2);
  h16* lwhT = (h16*)carve(96 * 32 * 2);
  h16* aw1T = (h16*)carve(256 * 128 * 2);
  h16* aw2T = (h16*)carve(256 * 256 * 2);
  h16* aw3T = (h16*)carve(32 * 256 * 2);
  h16* rw1T = (h16*)carve(256 * 32 * 2);
  h16* rw2T = (h16*)carve(256 * 256 * 2);
  (void)ws_size; (void)in_sizes; (void)n_in; (void)out_size;

  prep_weights<<<256, 256, 0, stream>>>(pgru_wx, pgru_wh, lgru_wx, lgru_wh,
                                        am_w1, am_w2, am_w3, ro_w1, ro_w2,
                                        pwxT, pwhT, lwxT, lwhT, aw1T, aw2T,
                                        aw3T, rw1T, rw2T);
  embed_kernel<<<(NLINKS + 255) / 256, 256, 0, stream>>>(
      capacity, scalev, le_w1, le_b1, le_w2, le_b2, lstate, lsf16, NLINKS);
  embed_kernel<<<(NPATHS + 255) / 256, 256, 0, stream>>>(
      traffic, packets, pe_w1, pe_b1, pe_w2, pe_b2, ips, (h16*)nullptr, NPATHS);
  init_seg<<<(NLINKS * 32 + 255) / 256, 256, 0, stream>>>(segsum, segmax,
                                                          segmin, cnt);
  count_links<<<(EE + 255) / 256, 256, 0, stream>>>(seq_links, cnt);

  const int path_blocks = (NPATHS + 127) / 128;  // 8 waves x 16 paths
  const int link_blocks = (NLINKS + 63) / 64;    // 4 waves x 16 links
  for (int it = 0; it < 8; ++it) {
    path_gru_kernel<<<path_blocks, 256, 0, stream>>>(
        lsf16, ips, l2p, pwxT, pwhT, pgru_bi, pgru_br, segsum, segmax, segmin);
    link_update_kernel<<<link_blocks, 128, 0, stream>>>(
        lstate, lsf16, segsum, segmax, segmin, cnt, aw1T, am_b1, aw2T, am_b2,
        aw3T, am_b3, lwxT, lwhT, lgru_bi, lgru_br);
  }
  readout_kernel<<<link_blocks, 128, 0, stream>>>(lsf16, rw1T, ro_b1, rw2T,
                                                  ro_b2, ro_w3, ro_b3, occ);
  final_delay_kernel<<<(NPATHS + 255) / 256, 256, 0, stream>>>(
      occ, capacity, scalev, l2p, (float*)d_out);
}
